// AggregatingUserEncoder_89429809038035
// MI455X (gfx1250) — matrix operand order repaired and checked by reading, not compiled, not run on hardware
//
#include <hip/hip_runtime.h>

// ---------------------------------------------------------------------------
// AggregatingUserEncoder for MI455X (gfx1250, wave32, WMMA, TDM).
// B=64, N=256, E=1024, H=16, D=64, S=257 (padded to 272).
// bf16 data / f32-accumulate WMMA for GEMMs; TDM double-buffered LDS staging
// of activations in the dominant QKV GEMM; fused scores+softmax in LDS with
// coalesced attn stores; only the CLS row propagated through out_proj/proj/LN.
// ---------------------------------------------------------------------------

constexpr int kB  = 64;
constexpr int kN  = 256;
constexpr int kE  = 1024;
constexpr int kH  = 16;
constexpr int kD  = 64;
constexpr int kS  = 257;   // N + 1 (CLS)
constexpr int kSP = 272;   // padded to multiple of 16
constexpr int kQT = 17;    // kSP / 16

typedef __attribute__((ext_vector_type(16))) __bf16 bf16x16;
typedef __attribute__((ext_vector_type(8)))  __bf16 bf16x8;
typedef __attribute__((ext_vector_type(8)))  float  f32x8;
typedef __attribute__((ext_vector_type(4)))  unsigned int u32x4;
typedef __attribute__((ext_vector_type(8)))  int i32x8;
typedef __attribute__((ext_vector_type(4)))  int i32x4;

#if defined(__HIP_DEVICE_COMPILE__) && \
    __has_builtin(__builtin_amdgcn_tensor_load_to_lds) && \
    __has_builtin(__builtin_amdgcn_s_wait_tensorcnt)
#define QKV_HAS_TDM 1
#else
#define QKV_HAS_TDM 0
#endif

// The TDM descriptor carries the LDS destination as an integer offset, so the
// optimizer cannot see that the TDM writes sA and would fold the staged-A
// ds_loads to undef ("no stores to sA exist"). Escape the LDS address into an
// asm with a memory clobber so the loads stay alive.
#if defined(__HIP_DEVICE_COMPILE__)
#define LDS_TDM_ESCAPE(p) asm volatile("" : : "v"((const void*)(p)) : "memory")
#else
#define LDS_TDM_ESCAPE(p) ((void)(p))
#endif

// ---------------------------------------------------------------------------
// 1) Build padded X = concat(cls, news) + pos_emb, cast to bf16.
// ---------------------------------------------------------------------------
__global__ void prep_x_kernel(const float* __restrict__ news,
                              const float* __restrict__ cls,
                              const float* __restrict__ pos,
                              __bf16* __restrict__ xbf) {
  int idx = blockIdx.x * blockDim.x + threadIdx.x;   // one float4 per thread
  int row = idx >> 8;                                // E/4 = 256 quads per row
  int e   = (idx & 255) << 2;
  int b = row / kSP, s = row % kSP;
  float v[4];
  if (s == 0) {
#pragma unroll
    for (int j = 0; j < 4; ++j) v[j] = cls[e + j] + pos[e + j];
  } else if (s < kS) {
    const float* nrow = news + ((size_t)(b * kN + (s - 1))) * kE + e;
    const float* prow = pos + (size_t)s * kE + e;
#pragma unroll
    for (int j = 0; j < 4; ++j) v[j] = nrow[j] + prow[j];
  } else {
#pragma unroll
    for (int j = 0; j < 4; ++j) v[j] = 0.0f;
  }
  __bf16* dst = xbf + (size_t)row * kE + e;
#pragma unroll
  for (int j = 0; j < 4; ++j) dst[j] = (__bf16)v[j];
}

// ---------------------------------------------------------------------------
// 2) f32 -> bf16 elementwise converter (weights, ctx).
// ---------------------------------------------------------------------------
__global__ void f32_to_bf16_kernel(const float* __restrict__ src,
                                   __bf16* __restrict__ dst, int n) {
  for (int i = blockIdx.x * blockDim.x + threadIdx.x; i < n;
       i += gridDim.x * blockDim.x)
    dst[i] = (__bf16)src[i];
}

// ---------------------------------------------------------------------------
// TDM: load a 64-row x 128-elem bf16 tile (row stride 1024 elems in memory)
// into LDS at byte offset lds_off, padding +16B per 256B row so the LDS row
// stride is 136 elems (16B-aligned ds_load_b128, conflict-free fragments).
// Descriptor layout per CDNA5 ISA.
// ---------------------------------------------------------------------------
#if QKV_HAS_TDM
__device__ inline void tdm_load_tile_64x128(const __bf16* gaddr, unsigned lds_off) {
  unsigned long long ga = (unsigned long long)(uintptr_t)gaddr;
  u32x4 g0;
  g0[0] = 1u;                                   // count=1 (user descriptor)
  g0[1] = lds_off;                              // lds_addr (sA is at offset 0)
  g0[2] = (unsigned)ga;                         // global_addr[31:0]
  g0[3] = (unsigned)((ga >> 32) & 0x1FFFFFFu)   // global_addr[56:32]
        | (2u << 30);                           // type = 2 (image/TDM)
  i32x8 g1;
  g1[0] = (1 << 16)    // data_size = 2 bytes (bf16)
        | (1 << 20)    // pad_enable
        | (5 << 22)    // pad_interval: every 64 DWORDs (one 256B row)
        | (3 << 25);   // pad_amount: 4 DWORDs (16B) -> LDS row = 272B
  g1[1] = (int)(128u << 16);   // tensor_dim0 = 128 elems
  g1[2] = (int)(64u << 16);    // tensor_dim1 = 64 rows
  g1[3] = (int)(128u << 16);   // tile_dim0 = 128
  g1[4] = 64;                  // tile_dim1 = 64, tile_dim2 = 0 (2D)
  g1[5] = 1024;                // tensor_dim0_stride = 1024 elems
  g1[6] = 0;
  g1[7] = 0;
  i32x4 z4 = {0, 0, 0, 0};
#if __has_include(<hip/amd_detail/amd_gfx1250_TDM.h>)
  i32x8 z8 = {0, 0, 0, 0, 0, 0, 0, 0};
  __builtin_amdgcn_tensor_load_to_lds(g0, g1, z4, z4, z8, 0);
#else
  __builtin_amdgcn_tensor_load_to_lds(g0, g1, z4, z4, 0);
#endif
}
#endif

// ---------------------------------------------------------------------------
// 3) QKV GEMM: out[m,n] = sum_k X[m,k]*Wqkv[n,k] + b[n].
//    Block tile 64(M) x 256(N); 8 waves, each wave a 32x64 register tile
//    (2x4 of 16x16 WMMA tiles -> 8 accumulators, 4 WMMAs per A-fragment).
//    A staged to LDS by TDM (double buffered, s_wait_tensorcnt), B streamed
//    from global (L2-resident, 6.3 MB).
// ---------------------------------------------------------------------------
__global__ void __launch_bounds__(256)
qkv_gemm_kernel(const __bf16* __restrict__ A,    // 17408 x 1024 bf16
                const __bf16* __restrict__ W,    // 3072 x 1024 bf16
                const float* __restrict__ bias,  // 3072
                __bf16* __restrict__ out) {      // 17408 x 3072 bf16
  constexpr int K = 1024, CH = 128, NCH = K / CH;   // 8 K-chunks
  constexpr int LROW = 136;                         // padded LDS row (elems)
  constexpr int LDO = 3 * kE;
  __shared__ __bf16 sA[2][64 * LROW];               // 2 x 17 KB

  const int bm = blockIdx.x;                // 272 blocks of 64 rows
  const int bn = blockIdx.y;                // 12 blocks of 256 cols
  const int tid = threadIdx.x, wave = tid >> 5, lane = tid & 31;
  const int mw = wave >> 2, nw = wave & 3;  // 2 x 4 wave grid
  const int lrow  = lane & 15;
  const int halfA = (lane >> 4) << 3;       // 0 / 8, unused: A frags use halfB
  const int halfB = (lane >> 4) << 4;       // B frag K offset 0 / 16
  const int row0 = bm * 64;
  const int col0 = bn * 256 + nw * 64;

  const __bf16* Wp0 = W + (size_t)(col0 + 0 * 16 + lrow) * K;
  const __bf16* Wp1 = W + (size_t)(col0 + 1 * 16 + lrow) * K;
  const __bf16* Wp2 = W + (size_t)(col0 + 2 * 16 + lrow) * K;
  const __bf16* Wp3 = W + (size_t)(col0 + 3 * 16 + lrow) * K;

  f32x8 acc[2][4] = {};

  auto compute_chunk = [&](const __bf16* sbuf, int c) {
#pragma unroll
    for (int kk = 0; kk < CH; kk += 32) {
      const int gk = c * CH + kk;
      __builtin_prefetch(Wp0 + gk + 128, 0, 1);
      __builtin_prefetch(Wp2 + gk + 128, 0, 1);
      bf16x16 b0 = *(const bf16x16*)(Wp0 + gk + halfB);
      bf16x16 b1 = *(const bf16x16*)(Wp1 + gk + halfB);
      bf16x16 b2 = *(const bf16x16*)(Wp2 + gk + halfB);
      bf16x16 b3 = *(const bf16x16*)(Wp3 + gk + halfB);
#pragma unroll
      for (int mi = 0; mi < 2; ++mi) {
        const __bf16* ap = sbuf + (mw * 32 + mi * 16 + lrow) * LROW + kk;
        bf16x8 lo = *(const bf16x8*)(ap + halfB);        // ds_load_b128
        bf16x8 hi = *(const bf16x8*)(ap + 8 + halfB);
        bf16x16 afr;
#pragma unroll
        for (int j = 0; j < 8; ++j) { afr[j] = lo[j]; afr[j + 8] = hi[j]; }
        acc[mi][0] = __builtin_amdgcn_wmma_f32_16x16x32_bf16(
            false, afr, false, b0, (short)0, acc[mi][0], false, false);
        acc[mi][1] = __builtin_amdgcn_wmma_f32_16x16x32_bf16(
            false, afr, false, b1, (short)0, acc[mi][1], false, false);
        acc[mi][2] = __builtin_amdgcn_wmma_f32_16x16x32_bf16(
            false, afr, false, b2, (short)0, acc[mi][2], false, false);
        acc[mi][3] = __builtin_amdgcn_wmma_f32_16x16x32_bf16(
            false, afr, false, b3, (short)0, acc[mi][3], false, false);
      }
    }
  };

#if QKV_HAS_TDM
  if (wave == 0) tdm_load_tile_64x128(A + (size_t)row0 * K, 0u);
  for (int c = 0; c < NCH; ++c) {
    if (wave == 0) {
      if (c + 1 < NCH) {
        tdm_load_tile_64x128(A + (size_t)row0 * K + (c + 1) * CH,
                             (unsigned)(((c + 1) & 1) * (64 * LROW * 2)));
        __builtin_amdgcn_s_wait_tensorcnt((short)1);   // chunk c landed
      } else {
        __builtin_amdgcn_s_wait_tensorcnt((short)0);
      }
      LDS_TDM_ESCAPE(&sA[0][0]);   // TDM wrote sA: keep ds_loads alive
    }
    __syncthreads();               // publish TDM-written LDS to all waves
    compute_chunk(sA[c & 1], c);
    __syncthreads();               // buffer reusable for chunk c+2
  }
#else
  for (int c = 0; c < NCH; ++c) {
    __syncthreads();
    for (int i = tid; i < 64 * 16; i += 256) {   // cooperative staging
      int r = i >> 4, g = (i & 15) << 3;
      *(bf16x8*)(&sA[0][r * LROW + g]) =
          *(const bf16x8*)(A + (size_t)(row0 + r) * K + c * CH + g);
    }
    __syncthreads();
    compute_chunk(sA[0], c);
  }
#endif

  // epilogue: C layout -> lanes>=16 hold rows +8; add bias, store bf16
#pragma unroll
  for (int mi = 0; mi < 2; ++mi) {
    int rbase = row0 + mw * 32 + mi * 16 + ((lane >> 4) << 3);
#pragma unroll
    for (int j = 0; j < 4; ++j) {
      int col = col0 + j * 16 + lrow;
      float bv = bias[col];
#pragma unroll
      for (int i = 0; i < 8; ++i)
        out[(size_t)(rbase + i) * LDO + col] = (__bf16)(acc[mi][j][i] + bv);
    }
  }
}

// ---------------------------------------------------------------------------
// 4) Generic small GEMM (out_proj / proj, 64 rows): one 16x16 tile per wave.
// ---------------------------------------------------------------------------
__global__ void gemm_bf16_wmma_kernel(const __bf16* __restrict__ A, int lda,
                                      const __bf16* __restrict__ W,
                                      const float* __restrict__ bias,
                                      __bf16* __restrict__ outb,
                                      float*  __restrict__ outf,
                                      int ldo, int Mtiles, int Ntiles, int K) {
  int w = blockIdx.x * (blockDim.x >> 5) + (threadIdx.x >> 5);
  if (w >= Mtiles * Ntiles) return;                 // wave-uniform
  int nt = w % Ntiles, mt = w / Ntiles;
  int lane  = threadIdx.x & 31;
  int lrow  = lane & 15;
  int halfA = (lane >> 4) << 3;
  int halfB = (lane >> 4) << 4;

  const __bf16* Ap = A + (size_t)(mt * 16 + lrow) * lda;
  const __bf16* Wp = W + (size_t)(nt * 16 + lrow) * K;

  f32x8 acc = {};
  for (int k = 0; k < K; k += 32) {
    __builtin_prefetch(Wp + k + 256, 0, 1);
    bf16x8  alo = *(const bf16x8*)(Ap + k + halfB);
    bf16x8  ahi = *(const bf16x8*)(Ap + k + 8 + halfB);
    bf16x16 bfr = *(const bf16x16*)(Wp + k + halfB);
    bf16x16 afr;
#pragma unroll
    for (int j = 0; j < 8; ++j) { afr[j] = alo[j]; afr[j + 8] = ahi[j]; }
    acc = __builtin_amdgcn_wmma_f32_16x16x32_bf16(false, afr, false, bfr,
                                                  (short)0, acc, false, false);
  }

  int col   = nt * 16 + lrow;
  int rbase = mt * 16 + ((lane >> 4) << 3);
  float bv  = bias ? bias[col] : 0.0f;
#pragma unroll
  for (int i = 0; i < 8; ++i) {
    float v = acc[i] + bv;
    size_t off = (size_t)(rbase + i) * ldo + col;
    if (outb) outb[off] = (__bf16)v;
    if (outf) outf[off] = v;
  }
}

// ---------------------------------------------------------------------------
// 5) Fused scores + masked softmax. One block = (b, h, q-tile of 16 rows),
//    4 waves; scores live in LDS (16 x 273 f32); attn written once with
//    fully coalesced row-sweep stores (128 contiguous floats / instruction).
// ---------------------------------------------------------------------------
__global__ void attn_softmax_kernel(const __bf16* __restrict__ qkv,
                                    const unsigned char* __restrict__ pmask,
                                    float* __restrict__ attn) {
  __shared__ float sc[16][273];
  __shared__ float red[16][8];
  __shared__ float rinv[16];

  int blk = blockIdx.x;
  int qt  = blk % kQT;
  int h   = (blk / kQT) % kH;
  int b   = blk / (kQT * kH);

  int tid  = threadIdx.x;         // 128 threads = 4 waves
  int wave = tid >> 5, lane = tid & 31;
  int lrow  = lane & 15;
  int halfA = (lane >> 4) << 3;
  int halfB = (lane >> 4) << 4;

  const __bf16* Qp = qkv + (size_t)(b * kSP + qt * 16 + lrow) * (3 * kE) + h * kD;
  const __bf16* Kb = qkv + (size_t)(b * kSP) * (3 * kE) + kE + h * kD;

  for (int kt = wave; kt < kQT; kt += 4) {
    f32x8 acc = {};
    const __bf16* Kp = Kb + (size_t)(kt * 16 + lrow) * (3 * kE);
#pragma unroll
    for (int d0 = 0; d0 < kD; d0 += 32) {
      bf16x8  alo = *(const bf16x8*)(Qp + d0 + halfB);
      bf16x8  ahi = *(const bf16x8*)(Qp + d0 + 8 + halfB);
      bf16x16 bfr = *(const bf16x16*)(Kp + d0 + halfB);
      bf16x16 afr;
#pragma unroll
      for (int j = 0; j < 8; ++j) { afr[j] = alo[j]; afr[j + 8] = ahi[j]; }
      acc = __builtin_amdgcn_wmma_f32_16x16x32_bf16(false, afr, false, bfr,
                                                    (short)0, acc, false, false);
    }
    int rbase = (lane >> 4) << 3;
#pragma unroll
    for (int i = 0; i < 8; ++i)
      sc[rbase + i][kt * 16 + lrow] = acc[i] * 0.125f;   // 1/sqrt(64)
  }
  __syncthreads();

  // softmax reductions: 8 threads per row
  int row = tid >> 3, sub = tid & 7;
  float mx = -3.0e38f;
  for (int c = sub; c < kSP; c += 8) {
    bool masked = (c >= kS) || (c > 0 && pmask[b * kN + (c - 1)] != 0);
    float v = masked ? -1.0e9f : sc[row][c];
    if (masked) sc[row][c] = -1.0e9f;
    mx = fmaxf(mx, v);
  }
  red[row][sub] = mx;
  __syncthreads();
  float rmx = red[row][0];
#pragma unroll
  for (int j = 1; j < 8; ++j) rmx = fmaxf(rmx, red[row][j]);
  __syncthreads();

  float sum = 0.0f;
  for (int c = sub; c < kSP; c += 8) {
    float e = __expf(sc[row][c] - rmx);   // v_exp_f32
    sc[row][c] = e;
    sum += e;
  }
  red[row][sub] = sum;
  __syncthreads();
  if (sub == 0) {
    float rs = 0.0f;
#pragma unroll
    for (int j = 0; j < 8; ++j) rs += red[row][j];
    rinv[row] = 1.0f / rs;
  }
  __syncthreads();

  // coalesced store: sweep rows, 128 threads cover contiguous columns
  float* base = attn + ((size_t)(b * kH + h) * kS + (size_t)qt * 16) * kS;
#pragma unroll 1
  for (int r = 0; r < 16; ++r) {
    if (qt * 16 + r >= kS) break;
    float inv = rinv[r];
    float* dst = base + (size_t)r * kS;
    for (int c = tid; c < kS; c += 128)
      dst[c] = sc[r][c] * inv;
  }
}

// ---------------------------------------------------------------------------
// 6) ctx for the CLS row only.
// ---------------------------------------------------------------------------
__global__ void ctx_q0_kernel(const float* __restrict__ attn,
                              const __bf16* __restrict__ qkv,
                              float* __restrict__ ctx) {
  int b = blockIdx.x >> 4, h = blockIdx.x & 15;
  int d = threadIdx.x;   // 64
  const float*  arow = attn + (size_t)(b * kH + h) * (kS * kS);   // q = 0 row
  const __bf16* V    = qkv + (size_t)(b * kSP) * (3 * kE) + 2 * kE + h * kD + d;
  float s = 0.0f;
  for (int k = 0; k < kS; ++k)
    s += arow[k] * (float)V[(size_t)k * (3 * kE)];
  ctx[(size_t)b * kE + h * kD + d] = s;
}

// ---------------------------------------------------------------------------
// 7) LayerNorm + ReLU + L2 normalize on the 64 CLS rows -> user output.
// ---------------------------------------------------------------------------
__global__ void ln_user_kernel(const float* __restrict__ p,
                               const float* __restrict__ gamma,
                               const float* __restrict__ beta,
                               float* __restrict__ user) {
  __shared__ float red[256];
  int b = blockIdx.x, t = threadIdx.x;
  const float* row = p + (size_t)b * kE;

  float vals[4];
  float s = 0.0f;
#pragma unroll
  for (int j = 0; j < 4; ++j) { vals[j] = row[t + j * 256]; s += vals[j]; }
  red[t] = s; __syncthreads();
  for (int st = 128; st > 0; st >>= 1) { if (t < st) red[t] += red[t + st]; __syncthreads(); }
  float mean = red[0] * (1.0f / kE); __syncthreads();

  float v = 0.0f;
#pragma unroll
  for (int j = 0; j < 4; ++j) { float d = vals[j] - mean; v += d * d; }
  red[t] = v; __syncthreads();
  for (int st = 128; st > 0; st >>= 1) { if (t < st) red[t] += red[t + st]; __syncthreads(); }
  float rstd = rsqrtf(red[0] * (1.0f / kE) + 1e-5f); __syncthreads();

  float ss = 0.0f;
#pragma unroll
  for (int j = 0; j < 4; ++j) {
    int i = t + j * 256;
    float u = (vals[j] - mean) * rstd * gamma[i] + beta[i];
    u = fmaxf(u, 0.0f);
    vals[j] = u;
    ss += u * u;
  }
  red[t] = ss; __syncthreads();
  for (int st = 128; st > 0; st >>= 1) { if (t < st) red[t] += red[t + st]; __syncthreads(); }
  float inv = 1.0f / fmaxf(sqrtf(red[0]), 1e-12f);
#pragma unroll
  for (int j = 0; j < 4; ++j)
    user[(size_t)b * kE + t + j * 256] = vals[j] * inv;
}

// ---------------------------------------------------------------------------
// Launch. Inputs (setup_inputs order): news_embeds, padding_mask(bool->u8),
// cls_token, pos_emb, in_proj_w, in_proj_b, out_proj_w, out_proj_b, proj_w,
// proj_b, ln_gamma, ln_beta. Output: user (64*1024 f32) then attn
// (64*16*257*257 f32). Workspace use ~154 MB.
// ---------------------------------------------------------------------------
extern "C" void kernel_launch(void* const* d_in, const int* in_sizes, int n_in,
                              void* d_out, int out_size, void* d_ws, size_t ws_size,
                              hipStream_t stream) {
  const float* news  = (const float*)d_in[0];
  const unsigned char* pmask = (const unsigned char*)d_in[1];   // jnp bool = 1B
  const float* cls   = (const float*)d_in[2];
  const float* pos   = (const float*)d_in[3];
  const float* w_qkv = (const float*)d_in[4];
  const float* b_qkv = (const float*)d_in[5];
  const float* w_out = (const float*)d_in[6];
  const float* b_out = (const float*)d_in[7];
  const float* w_prj = (const float*)d_in[8];
  const float* b_prj = (const float*)d_in[9];
  const float* gamma = (const float*)d_in[10];
  const float* beta  = (const float*)d_in[11];

  char* wsp = (char*)d_ws;
  auto alloc = [&](size_t bytes) {
    char* p = wsp;
    wsp += (bytes + 255) & ~(size_t)255;
    return p;
  };
  __bf16* xbf   = (__bf16*)alloc((size_t)kB * kSP * kE * 2);        // 35.7 MB
  __bf16* wqkvb = (__bf16*)alloc((size_t)3 * kE * kE * 2);          //  6.3 MB
  __bf16* woutb = (__bf16*)alloc((size_t)kE * kE * 2);              //  2.1 MB
  __bf16* wprjb = (__bf16*)alloc((size_t)kE * kE * 2);              //  2.1 MB
  __bf16* qkvb  = (__bf16*)alloc((size_t)kB * kSP * 3 * kE * 2);    // 107  MB
  float*  ctx   = (float*) alloc((size_t)kB * kE * 4);
  __bf16* ctxb  = (__bf16*)alloc((size_t)kB * kE * 2);
  __bf16* aoutb = (__bf16*)alloc((size_t)kB * kE * 2);
  float*  pbuf  = (float*) alloc((size_t)kB * kE * 4);

  float* user = (float*)d_out;
  float* attn = user + (size_t)kB * kE;

  // 1) padded bf16 X
  prep_x_kernel<<<kB * kSP, 256, 0, stream>>>(news, cls, pos, xbf);
  // 2) bf16 weights
  f32_to_bf16_kernel<<<2048, 256, 0, stream>>>(w_qkv, wqkvb, 3 * kE * kE);
  f32_to_bf16_kernel<<<2048, 256, 0, stream>>>(w_out, woutb, kE * kE);
  f32_to_bf16_kernel<<<2048, 256, 0, stream>>>(w_prj, wprjb, kE * kE);
  // 3) QKV GEMM: 272 x 12 blocks of 64x256, TDM-staged A, reg-blocked WMMA
  qkv_gemm_kernel<<<dim3(272, 12), 256, 0, stream>>>(xbf, wqkvb, b_qkv, qkvb);
  // 4) fused scores + softmax -> attn output (single 270 MB write)
  attn_softmax_kernel<<<kB * kH * kQT, 128, 0, stream>>>(qkvb, pmask, attn);
  // 5) ctx for CLS rows only
  ctx_q0_kernel<<<kB * kH, 64, 0, stream>>>(attn, qkvb, ctx);
  f32_to_bf16_kernel<<<256, 256, 0, stream>>>(ctx, ctxb, kB * kE);
  // 6) out_proj then proj (64 rows each): Mtiles=4, Ntiles=64
  gemm_bf16_wmma_kernel<<<32, 256, 0, stream>>>(
      ctxb, kE, woutb, b_out, aoutb, nullptr, kE, 4, 64, kE);
  gemm_bf16_wmma_kernel<<<32, 256, 0, stream>>>(
      aoutb, kE, wprjb, b_prj, nullptr, pbuf, kE, 4, 64, kE);
  // 7) LN + ReLU + L2 normalize -> user
  ln_user_kernel<<<kB, 256, 0, stream>>>(pbuf, gamma, beta, user);
}
